// ContinusConvolution_24043226923188
// MI455X (gfx1250) — matrix operand order repaired and checked by reading, not compiled, not run on hardware
//
#include <hip/hip_runtime.h>

// Problem constants (from the reference)
#define B_   4
#define N_   1024
#define NB_  32
#define C_   384
#define CZ_  128
#define PAIRS (B_ * N_)          // 4096
#define GRID1 512                // kernel-1 workgroups (2 col-blocks x 256 pair-streams)
#define PSTRIDE 256              // pair stride per workgroup

typedef __attribute__((ext_vector_type(16))) __bf16   bf16x16;
typedef __attribute__((ext_vector_type(8)))  float    f32x8;
typedef __attribute__((ext_vector_type(2)))  float    f32x2;
typedef __attribute__((ext_vector_type(8)))  unsigned uint32x8;
typedef __attribute__((ext_vector_type(4)))  unsigned uint32x4;

// f32 -> bf16 (round-to-nearest-even), packed pair into one dword
static __device__ __forceinline__ unsigned bf_bits(float f) {
  unsigned u = __builtin_bit_cast(unsigned, f);
  return (u + 0x7FFFu + ((u >> 16) & 1u)) >> 16;
}
static __device__ __forceinline__ unsigned pk_bf16(float lo, float hi) {
  return bf_bits(lo) | (bf_bits(hi) << 16);
}

// Fragment-packed LDS layout: frag = 256 dwords = 64 x 16B chunks.
// Lane l owns chunks 2l, 2l+1, XOR-swizzled so lanes {0,8,16,24} hit distinct banks.
static __device__ __forceinline__ int frag_off(int frag, int lane, int j) {
  int chunk = lane * 2 + (j >> 2);
  chunk ^= ((lane >> 3) & 3) << 2;          // bank swizzle (16B granularity)
  return frag * 256 + chunk * 4 + (j & 3);  // dword index
}

// Load one 16x32-bf16 A/B fragment: two ds_load_b128, conflict-free.
static __device__ __forceinline__ bf16x16
load_frag(const unsigned* s, int frag, int lane) {
  int c0 = (lane * 2) ^ (((lane >> 3) & 3) << 2);
  const uint32x4* p = (const uint32x4*)(s + frag * 256 + c0 * 4);
  uint32x4 a = p[0], b = p[1];              // chunk c0, c0+1 (== (2l+1)^swz)
  uint32x8 u = {a[0], a[1], a[2], a[3], b[0], b[1], b[2], b[3]};
  return __builtin_bit_cast(bf16x16, u);
}

static __device__ __forceinline__ f32x8 wmma_bf16(bf16x16 a, bf16x16 b, f32x8 c) {
  // (neg_a, A, neg_b, B, c_mod, C, reuse_a, reuse_b)
  return __builtin_amdgcn_wmma_f32_16x16x32_bf16(false, a, false, b, (short)0, c,
                                                 false, false);
}

// Decompose a linear packed-A dword index into (frag, lane, j) and source coords.
// A-fragment packing used here (16-bit A 16x32, contiguous k, same as B): lane holds row (lane&15);
// K = (lane&16 ? 16:0) + 2j + 32*ks.
static __device__ __forceinline__ void
a_coords(int idx, int nfragk, int rowstride, int& off, long& src) {
  int j = idx & 7, ln = (idx >> 3) & 31, f = idx >> 8;
  int mt = f / nfragk, ks = f - mt * nfragk;
  int row = mt * 16 + (ln & 15);
  int kb  = ks * 32 + ((ln & 16) ? 16 : 0) + 2 * j;
  off = frag_off(f, ln, j);
  src = (long)row * rowstride + kb;
}

// Kernel 1: s_new(pre-LN) = sum_k m_k * (s_ij@Ws - s_i) * (z_ij@Wz)
// Workgroup: fixed 192-column block (cb), streams PAIRS/PSTRIDE pairs,
// register-prefetch double buffering of the per-pair activation tiles.
__global__ __launch_bounds__(384)
void msg_kernel(const float* __restrict__ s_i, const float* __restrict__ s_ij,
                const float* __restrict__ m_ij, const float* __restrict__ z_ij,
                const float* __restrict__ Ws, const float* __restrict__ Wz,
                float* __restrict__ out) {
  __shared__ __align__(16) unsigned sWs[12 * 12 * 256];  // 147456 B, frag-packed bf16
  __shared__ __align__(16) unsigned sWz[12 * 4 * 256];   //  49152 B, frag-packed bf16
  __shared__ __align__(16) unsigned sA[24 * 256];        //  24576 B, frag-packed bf16
  __shared__ __align__(16) unsigned sZ[8 * 256];         //   8192 B, frag-packed bf16
  __shared__ float sSi[192];
  __shared__ float sM[32];

  const int tid   = threadIdx.x;
  const int cb    = blockIdx.x & 1;    // column block: [cb*192, cb*192+192)
  const int p0    = blockIdx.x >> 1;   // first pair
  const int lane  = tid & 31;
  const int w     = tid >> 5;          // wave id == local N-tile (0..11)
  const int ncol0 = cb * 192;

  // ---- one-time: gather Ws/Wz slice (L2-resident), pack B-fragments into LDS ----
  // B-fragment map (16-bit B 32x16): lane holds col (lane&15); K = (lane&16?16:0)+2j.
  for (int idx = tid; idx < 12 * 12 * 256; idx += 384) {
    int j = idx & 7, ln = (idx >> 3) & 31, t = idx >> 8;  // t = nt*12 + ks
    int ks = t % 12, nt = t / 12;
    int n  = ncol0 + nt * 16 + (ln & 15);
    int k0 = ks * 32 + ((ln & 16) ? 16 : 0) + 2 * j;
    sWs[frag_off(t, ln, j)] =
        pk_bf16(Ws[(size_t)k0 * C_ + n], Ws[(size_t)(k0 + 1) * C_ + n]);
  }
  for (int idx = tid; idx < 12 * 4 * 256; idx += 384) {
    int j = idx & 7, ln = (idx >> 3) & 31, t = idx >> 8;  // t = nt*4 + ks
    int ks = t & 3, nt = t >> 2;
    int n  = ncol0 + nt * 16 + (ln & 15);
    int k0 = ks * 32 + ((ln & 16) ? 16 : 0) + 2 * j;
    sWz[frag_off(t, ln, j)] =
        pk_bf16(Wz[(size_t)k0 * C_ + n], Wz[(size_t)(k0 + 1) * C_ + n]);
  }

  // ---- register prefetch buffers (next pair's tiles) ----
  f32x2 rA[16];   // 24*256 dwords / 384 thr = 16 f32x2 each
  f32x2 rZ[6];    //  8*256 dwords / 384 thr = 5.33 -> 6 slots, last partial
  float rSi = 0.f, rM = 0.f;
  int   offA[16], offZ[6];
  long  srcA[16], srcZ[6];
#pragma unroll
  for (int k = 0; k < 16; ++k) a_coords(tid + k * 384, 12, C_, offA[k], srcA[k]);
#pragma unroll
  for (int k = 0; k < 6; ++k) {
    int idx = tid + k * 384;
    if (idx < 8 * 256) a_coords(idx, 4, CZ_, offZ[k], srcZ[k]);
    else { offZ[k] = -1; srcZ[k] = 0; }
  }

  // prefetch first pair
  {
    const float* As = s_ij + (size_t)p0 * (NB_ * C_);
    const float* Zs = z_ij + (size_t)p0 * (NB_ * CZ_);
#pragma unroll
    for (int k = 0; k < 16; ++k) rA[k] = *(const f32x2*)(As + srcA[k]);
#pragma unroll
    for (int k = 0; k < 6; ++k)
      if (offZ[k] >= 0) rZ[k] = *(const f32x2*)(Zs + srcZ[k]);
    if (tid < 192) rSi = s_i[(size_t)p0 * C_ + ncol0 + tid];
    if (tid < 32)  rM  = m_ij[(size_t)p0 * NB_ + tid];
  }

  for (int pair = p0; pair < PAIRS; pair += PSTRIDE) {
    __syncthreads();  // previous iteration's LDS readers are done

    // ---- commit prefetched tiles: f32 -> bf16, fragment-packed stores ----
#pragma unroll
    for (int k = 0; k < 16; ++k) sA[offA[k]] = pk_bf16(rA[k][0], rA[k][1]);
#pragma unroll
    for (int k = 0; k < 6; ++k)
      if (offZ[k] >= 0) sZ[offZ[k]] = pk_bf16(rZ[k][0], rZ[k][1]);
    if (tid < 192) sSi[tid] = rSi;
    if (tid < 32)  sM[tid]  = rM;
    __syncthreads();

    // ---- issue next pair's global loads (overlap with WMMA below) ----
    const int nxt = pair + PSTRIDE;
    if (nxt < PAIRS) {
      const float* As = s_ij + (size_t)nxt * (NB_ * C_);
      const float* Zs = z_ij + (size_t)nxt * (NB_ * CZ_);
#pragma unroll
      for (int k = 0; k < 16; ++k) rA[k] = *(const f32x2*)(As + srcA[k]);
#pragma unroll
      for (int k = 0; k < 6; ++k)
        if (offZ[k] >= 0) rZ[k] = *(const f32x2*)(Zs + srcZ[k]);
      if (tid < 192) rSi = s_i[(size_t)nxt * C_ + ncol0 + tid];
      if (tid < 32)  rM  = m_ij[(size_t)nxt * NB_ + tid];
    }

    // ---- WMMA: proj = A @ Ws, gate = Z @ Wz (this wave's 16 output columns) ----
    f32x8 accP0 = {0, 0, 0, 0, 0, 0, 0, 0};
    f32x8 accP1 = accP0, accG0 = accP0, accG1 = accP0;
#pragma unroll
    for (int ks = 0; ks < 12; ++ks) {  // K = 384
      bf16x16 b  = load_frag(sWs, w * 12 + ks, lane);
      bf16x16 a0 = load_frag(sA, ks, lane);        // mt=0 -> frag ks
      bf16x16 a1 = load_frag(sA, 12 + ks, lane);   // mt=1 -> frag 12+ks
      accP0 = wmma_bf16(a0, b, accP0);
      accP1 = wmma_bf16(a1, b, accP1);
    }
#pragma unroll
    for (int ks = 0; ks < 4; ++ks) {   // K = 128
      bf16x16 b  = load_frag(sWz, w * 4 + ks, lane);
      bf16x16 a0 = load_frag(sZ, ks, lane);
      bf16x16 a1 = load_frag(sZ, 4 + ks, lane);
      accG0 = wmma_bf16(a0, b, accG0);
      accG1 = wmma_bf16(a1, b, accG1);
    }

    // ---- epilogue: (proj - s_i) * gate * mask, reduce over 32 neighbors ----
    // C/D layout: VGPR r, lane l -> element [M = r + (l<16?0:8)][N = l&15]
    float si = sSi[w * 16 + (lane & 15)];
    int   hi = (lane & 16) ? 8 : 0;
    float sum = 0.f;
#pragma unroll
    for (int r = 0; r < 8; ++r) {
      sum += (accP0[r] - si) * accG0[r] * sM[r + hi];       // neighbors 0..15
      sum += (accP1[r] - si) * accG1[r] * sM[16 + r + hi];  // neighbors 16..31
    }
    sum += __shfl_xor(sum, 16, 32);  // combine lane-halves (rows M and M+8)
    if (lane < 16)
      out[(size_t)pair * C_ + ncol0 + w * 16 + lane] = sum;
  }
}

// Kernel 2: in-place layernorm over C=384 per row.
__global__ __launch_bounds__(128)
void ln_kernel(float* __restrict__ out, const float* __restrict__ gamma,
               const float* __restrict__ beta) {
  __shared__ float rs[128], rq[128];
  const int row = blockIdx.x, t = threadIdx.x;
  float* p = out + (size_t)row * C_;
  float x0 = p[t], x1 = p[t + 128], x2 = p[t + 256];
  rs[t] = x0 + x1 + x2;
  rq[t] = x0 * x0 + x1 * x1 + x2 * x2;
  __syncthreads();
#pragma unroll
  for (int off = 64; off > 0; off >>= 1) {
    if (t < off) { rs[t] += rs[t + off]; rq[t] += rq[t + off]; }
    __syncthreads();
  }
  float mu  = rs[0] * (1.0f / C_);
  float var = rq[0] * (1.0f / C_) - mu * mu;
  float inv = rsqrtf(var + 1e-6f);
  p[t]       = (x0 - mu) * inv * gamma[t]       + beta[t];
  p[t + 128] = (x1 - mu) * inv * gamma[t + 128] + beta[t + 128];
  p[t + 256] = (x2 - mu) * inv * gamma[t + 256] + beta[t + 256];
}

extern "C" void kernel_launch(void* const* d_in, const int* in_sizes, int n_in,
                              void* d_out, int out_size, void* d_ws, size_t ws_size,
                              hipStream_t stream) {
  (void)in_sizes; (void)n_in; (void)out_size; (void)d_ws; (void)ws_size;
  const float* s_i   = (const float*)d_in[0];
  const float* s_ij  = (const float*)d_in[1];
  const float* m_ij  = (const float*)d_in[2];
  const float* z_ij  = (const float*)d_in[3];
  const float* W_s   = (const float*)d_in[4];
  const float* W_z   = (const float*)d_in[5];
  const float* gamma = (const float*)d_in[6];
  const float* beta  = (const float*)d_in[7];
  float* out = (float*)d_out;

  msg_kernel<<<GRID1, 384, 0, stream>>>(s_i, s_ij, m_ij, z_ij, W_s, W_z, out);
  ln_kernel<<<PAIRS, 128, 0, stream>>>(out, gamma, beta);
}
